// GAT_22686017257932
// MI455X (gfx1250) — matrix operand order repaired and checked by reading, not compiled, not run on hardware
//
#include <hip/hip_runtime.h>
#include <hip/hip_bf16.h>

// ---------------- problem constants ----------------
#define NN      50000
#define EE      800000
#define NE      850000      /* E + N self loops */
#define F_IN    64
#define HID     32
#define HEADS   4
#define DLAY    (HEADS*HID) /* 128 */
#define NCLS    8
#define BN_EPS  1e-5f
#define SLOPE   0.2f

typedef __attribute__((ext_vector_type(16))) _Float16     v16h;
typedef __attribute__((ext_vector_type(8)))  float        v8f;
typedef __attribute__((ext_vector_type(4)))  unsigned int v4u;

// ---------------- small utility kernels ----------------

__global__ void gat_fill_k(float* __restrict__ p, float v, int n) {
  int i = blockIdx.x * blockDim.x + threadIdx.x;
  if (i < n) p[i] = v;
}

// elementwise: v = in + bias(optional); v = BN(v); v = ELU(v) (optional);
// write f16 and/or f32
__global__ void gat_ew_bn_k(const float* __restrict__ in, int D, int total,
                            const float* __restrict__ bias,
                            const float* __restrict__ gamma, const float* __restrict__ beta,
                            const float* __restrict__ mean,  const float* __restrict__ var,
                            int do_elu,
                            _Float16* __restrict__ out16, float* __restrict__ out32) {
  int i = blockIdx.x * blockDim.x + threadIdx.x;
  if (i >= total) return;
  int d = i % D;
  float v = in[i];
  if (bias) v += bias[d];
  v = (v - mean[d]) * (gamma[d] * rsqrtf(var[d] + BN_EPS)) + beta[d];
  if (do_elu) v = (v > 0.0f) ? v : (expf(v) - 1.0f);
  if (out16) out16[i] = (_Float16)v;
  if (out32) out32[i] = v;
}

// ---------------- weight pre-pack into WMMA B-fragment order ----------------
// B fragment layout (f16 32x16, CDNA5 ISA 7.12.2): lane L supplies column n = L&15,
// lanes 0-15 hold K 0..7,16..23, lanes 16-31 hold K 8..15,24..31, packed 2 f16 per dword.
// Packed layout: halves[((nt*KB + kb)*32 + lane)*16 + 2*i {,+1}]
// so the GEMM reads each lane's fragment as two contiguous b128 loads.
__global__ void gat_pack_b_k(const float* __restrict__ W, int ncols_in,
                             _Float16* __restrict__ Bp, int K, int NT) {
  int idx = blockIdx.x * blockDim.x + threadIdx.x;
  int KB = K / 32;
  int total = NT * KB * 32 * 8;          // one thread per packed dword
  if (idx >= total) return;
  int i    = idx & 7;
  int lane = (idx >> 3) & 31;
  int kb   = (idx >> 8) % KB;
  int nt   = (idx >> 8) / KB;
  int hi = lane >> 4, lr = lane & 15;
  int col = nt * 16 + lr;
  int k0  = kb * 32 + 8 * hi + 2 * i + 8 * (i >> 2);
  _Float16 e0 = (col < ncols_in) ? (_Float16)W[(size_t)k0       * ncols_in + col] : (_Float16)0.0f;
  _Float16 e1 = (col < ncols_in) ? (_Float16)W[(size_t)(k0 + 1) * ncols_in + col] : (_Float16)0.0f;
  Bp[(size_t)idx * 2]     = e0;
  Bp[(size_t)idx * 2 + 1] = e1;
}

// ---------------- WMMA GEMM ----------------
// C[f32, M x 16*NT] = A[f16, M x K] @ B(packed fragments)
// One wave per 16-row stripe; the A fragment is loaded once per K-step (two b128
// loads/lane) and reused across all NT column tiles; B fragments are two b128
// loads/lane from the packed layout. Fully unrolled -> NT*K/32 v_wmma per kernel.
// A 16x32 f16 layout: lanes 0-15 = rows; K 0..7 (lo half) / 8..15 (hi half) then +16.
// C 16x16 f32 layout: vgpr r holds row r (lanes 0-15) / row 8+r (lanes 16-31).
template <int K, int NT>
__global__ void gat_gemm_wmma_k(const _Float16* __restrict__ A, int lda,
                                const _Float16* __restrict__ Bp,
                                float* __restrict__ C, int ldc) {
  constexpr int KB = K / 32;
  const int mt   = blockIdx.x;
  const int lane = threadIdx.x & 31;
  const int hi   = lane >> 4;
  const int lr   = lane & 15;
  const int mrow = mt * 16 + lr;

  v8f acc[NT];
#pragma unroll
  for (int nt = 0; nt < NT; ++nt) acc[nt] = (v8f){};

#pragma unroll
  for (int kb = 0; kb < KB; ++kb) {
    union { v4u q[2]; v16h h; } ua;
    const _Float16* arow = A + (size_t)mrow * lda + kb * 32 + 8 * hi;
    ua.q[0] = *(const v4u*)(arow);        // K group base, 8 f16 = 16B contiguous
    ua.q[1] = *(const v4u*)(arow + 16);   // +16 K group
#pragma unroll
    for (int nt = 0; nt < NT; ++nt) {
      union { v4u q[2]; v16h h; } ub;
      const _Float16* bp = Bp + ((size_t)(nt * KB + kb) * 32 + lane) * 16;
      ub.q[0] = *(const v4u*)(bp);
      ub.q[1] = *(const v4u*)(bp + 8);
      acc[nt] = __builtin_amdgcn_wmma_f32_16x16x32_f16(
          /*neg_a=*/false, ua.h, /*neg_b=*/false, ub.h,
          /*c_mod=*/(short)0, acc[nt], /*reuse_a=*/false, /*reuse_b=*/false);
    }
  }

#pragma unroll
  for (int nt = 0; nt < NT; ++nt) {
#pragma unroll
    for (int r = 0; r < 8; ++r) {
      int m = mt * 16 + r + 8 * hi;
      C[(size_t)m * ldc + nt * 16 + lr] = acc[nt][r];
    }
  }
}

// ---------------- attention scores ----------------
__global__ void gat_att_k(const float* __restrict__ hW, int ld, int H, int C,
                          const float* __restrict__ att_src, const float* __restrict__ att_dst,
                          float* __restrict__ asrc, float* __restrict__ adst, int nH) {
  int i = blockIdx.x * blockDim.x + threadIdx.x;
  if (i >= nH) return;
  int n = i / H, h = i % H;
  const float* row = hW + (size_t)n * ld + h * C;
  float s = 0.0f, d = 0.0f;
  for (int c = 0; c < C; ++c) {
    float v = row[c];
    s += v * att_src[h * C + c];
    d += v * att_dst[h * C + c];
  }
  asrc[i] = s;
  adst[i] = d;
}

// ---------------- edge kernels ----------------
__device__ __forceinline__ void gat_edge_sd(const int* __restrict__ ei, int e, int& s, int& d) {
  if (e < EE) { s = ei[e]; d = ei[EE + e]; }
  else        { s = e - EE; d = e - EE; }        // self loops
}

__device__ __forceinline__ void gat_atomic_max_f32(float* addr, float val) {
  unsigned int* ua = (unsigned int*)addr;
  unsigned int old = *ua;
  while (__uint_as_float(old) < val) {
    unsigned int assumed = old;
    old = atomicCAS(ua, assumed, __float_as_uint(val));
    if (old == assumed) break;
  }
}

// pass 1: e = leaky_relu(asrc[src]+adst[dst]); segment max over dst
__global__ void gat_edge1_k(const int* __restrict__ ei,
                            const float* __restrict__ asrc, const float* __restrict__ adst,
                            float* __restrict__ emax, float* __restrict__ edge_e, int H) {
  int i = blockIdx.x * blockDim.x + threadIdx.x;
  if (i >= NE * H) return;
  int e = i / H, h = i % H;
  int s, d; gat_edge_sd(ei, e, s, d);
  float v = asrc[s * H + h] + adst[d * H + h];
  v = (v > 0.0f) ? v : SLOPE * v;
  edge_e[i] = v;
  gat_atomic_max_f32(&emax[d * H + h], v);
}

// pass 2: ex = exp(e - max); segment sum of ex over dst
__global__ void gat_edge2_k(const int* __restrict__ ei,
                            const float* __restrict__ emax,
                            float* __restrict__ edge_e, float* __restrict__ denom, int H) {
  int i = blockIdx.x * blockDim.x + threadIdx.x;
  if (i >= NE * H) return;
  int e = i / H, h = i % H;
  int s, d; gat_edge_sd(ei, e, s, d);
  float ex = expf(edge_e[i] - emax[d * H + h]);
  edge_e[i] = ex;
  atomicAdd(&denom[d * H + h], ex);
}

// pass 3: agg[dst, j] += hW[src, j] * alpha   (j = h*C+c, D = H*C)
__global__ void gat_edge3_k(const int* __restrict__ ei,
                            const float* __restrict__ hW, int ld,
                            const float* __restrict__ edge_e, const float* __restrict__ denom,
                            float* __restrict__ agg, int H, int C, int D) {
  int i = blockIdx.x * blockDim.x + threadIdx.x;
  if (i >= NE * D) return;
  int e = i / D, j = i % D;
  int h = j / C;
  int s, d; gat_edge_sd(ei, e, s, d);
  float alpha = edge_e[e * H + h] / (denom[d * H + h] + 1e-16f);
  atomicAdd(&agg[(size_t)d * D + j], hW[(size_t)s * ld + j] * alpha);
}

// ---------------- final log-softmax ----------------
__global__ void gat_logsoftmax_k(const float* __restrict__ agg, const float* __restrict__ bias,
                                 float* __restrict__ out) {
  int n = blockIdx.x * blockDim.x + threadIdx.x;
  if (n >= NN) return;
  float v[NCLS];
  float mx = -1e30f;
#pragma unroll
  for (int c = 0; c < NCLS; ++c) {
    v[c] = agg[n * NCLS + c] + bias[c];
    mx = fmaxf(mx, v[c]);
  }
  float sum = 0.0f;
#pragma unroll
  for (int c = 0; c < NCLS; ++c) sum += expf(v[c] - mx);
  float lse = mx + logf(sum);
#pragma unroll
  for (int c = 0; c < NCLS; ++c) out[n * NCLS + c] = v[c] - lse;
}

// ---------------- launch ----------------
static inline int gat_cdiv(int a, int b) { return (a + b - 1) / b; }

extern "C" void kernel_launch(void* const* d_in, const int* in_sizes, int n_in,
                              void* d_out, int out_size, void* d_ws, size_t ws_size,
                              hipStream_t stream) {
  (void)in_sizes; (void)n_in; (void)out_size; (void)ws_size;
  // ---- inputs (setup_inputs dict order, params flattened depth-first in insertion order)
  const float* x      = (const float*)d_in[0];
  const int*   ei     = (const int*)  d_in[1];
  const float* bnin_g = (const float*)d_in[2];
  const float* bnin_b = (const float*)d_in[3];
  const float* bnin_m = (const float*)d_in[4];
  const float* bnin_v = (const float*)d_in[5];
  const float* projW  = (const float*)d_in[6];
  const float* projb  = (const float*)d_in[7];
  const float* bnp_g  = (const float*)d_in[8];
  const float* bnp_b  = (const float*)d_in[9];
  const float* bnp_m  = (const float*)d_in[10];
  const float* bnp_v  = (const float*)d_in[11];
  const float* convW [2] = {(const float*)d_in[12], (const float*)d_in[16]};
  const float* convAS[2] = {(const float*)d_in[13], (const float*)d_in[17]};
  const float* convAD[2] = {(const float*)d_in[14], (const float*)d_in[18]};
  const float* convB [2] = {(const float*)d_in[15], (const float*)d_in[19]};
  const float* bn_g[2] = {(const float*)d_in[20], (const float*)d_in[24]};
  const float* bn_b[2] = {(const float*)d_in[21], (const float*)d_in[25]};
  const float* bn_m[2] = {(const float*)d_in[22], (const float*)d_in[26]};
  const float* bn_v[2] = {(const float*)d_in[23], (const float*)d_in[27]};
  const float* outW  = (const float*)d_in[28];
  const float* outAS = (const float*)d_in[29];
  const float* outAD = (const float*)d_in[30];
  const float* outB  = (const float*)d_in[31];
  float* out = (float*)d_out;

  // ---- workspace carve (256B aligned)
  char* cur = (char*)d_ws;
  auto carve = [&](size_t bytes) -> void* {
    void* p = (void*)cur;
    cur += (bytes + 255) & ~(size_t)255;
    return p;
  };
  _Float16* wprojP  = (_Float16*)carve((size_t)F_IN * HID * 2);        // packed 64x32
  _Float16* w0P     = (_Float16*)carve((size_t)HID * DLAY * 2);        // packed 32x128
  _Float16* w1P     = (_Float16*)carve((size_t)DLAY * DLAY * 2);       // packed 128x128
  _Float16* woutP   = (_Float16*)carve((size_t)DLAY * 16 * 2);         // packed 128x16 (8 padded)
  _Float16* xh16    = (_Float16*)carve((size_t)NN * F_IN * 2);
  _Float16* act16   = (_Float16*)carve((size_t)NN * DLAY * 2);
  float*    hW      = (float*)   carve((size_t)NN * DLAY * 4);
  float*    agg     = (float*)   carve((size_t)NN * DLAY * 4);
  float*    asrc    = (float*)   carve((size_t)NN * HEADS * 4);
  float*    adst    = (float*)   carve((size_t)NN * HEADS * 4);
  float*    emax    = (float*)   carve((size_t)NN * HEADS * 4);
  float*    denom   = (float*)   carve((size_t)NN * HEADS * 4);
  float*    edge_e  = (float*)   carve((size_t)NE * HEADS * 4);

  const int B256 = 256;
  const int MT = NN / 16;   // 3125 exact row tiles

  // ---- pack weights into WMMA fragment order (one thread per packed dword)
  gat_pack_b_k<<<gat_cdiv(2 * 2 * 256, B256), B256, 0, stream>>>(projW,    HID,  wprojP, F_IN, 2);
  gat_pack_b_k<<<gat_cdiv(8 * 1 * 256, B256), B256, 0, stream>>>(convW[0], DLAY, w0P,    HID,  8);
  gat_pack_b_k<<<gat_cdiv(8 * 4 * 256, B256), B256, 0, stream>>>(convW[1], DLAY, w1P,    DLAY, 8);
  gat_pack_b_k<<<gat_cdiv(1 * 4 * 256, B256), B256, 0, stream>>>(outW,     NCLS, woutP,  DLAY, 1);

  // ---- input BN -> f16
  gat_ew_bn_k<<<gat_cdiv(NN * F_IN, B256), B256, 0, stream>>>(
      x, F_IN, NN * F_IN, nullptr, bnin_g, bnin_b, bnin_m, bnin_v, 0, xh16, nullptr);

  // ---- proj GEMM (WMMA): hW[N,32] = xh16[N,64] @ projW[64,32]
  gat_gemm_wmma_k<F_IN, 2><<<MT, 32, 0, stream>>>(xh16, F_IN, wprojP, hW, HID);

  // ---- + proj_b, BN, ELU -> act16[N,32]
  gat_ew_bn_k<<<gat_cdiv(NN * HID, B256), B256, 0, stream>>>(
      hW, HID, NN * HID, projb, bnp_g, bnp_b, bnp_m, bnp_v, 1, act16, nullptr);

  // ---- two GAT layers (4 heads x 32, concat) ----
  for (int l = 0; l < 2; ++l) {
    // hW[N,128] = act16[N,K] @ W[K,128]
    if (l == 0) gat_gemm_wmma_k<HID,  8><<<MT, 32, 0, stream>>>(act16, HID,  w0P, hW, DLAY);
    else        gat_gemm_wmma_k<DLAY, 8><<<MT, 32, 0, stream>>>(act16, DLAY, w1P, hW, DLAY);

    // attention scores
    gat_att_k<<<gat_cdiv(NN * HEADS, B256), B256, 0, stream>>>(
        hW, DLAY, HEADS, HID, convAS[l], convAD[l], asrc, adst, NN * HEADS);

    // init reductions + aggregation
    gat_fill_k<<<gat_cdiv(NN * HEADS, B256), B256, 0, stream>>>(emax,  -1e30f, NN * HEADS);
    gat_fill_k<<<gat_cdiv(NN * HEADS, B256), B256, 0, stream>>>(denom,  0.0f,  NN * HEADS);
    gat_fill_k<<<gat_cdiv(NN * DLAY,  B256), B256, 0, stream>>>(agg,    0.0f,  NN * DLAY);

    // segment softmax + message aggregation
    gat_edge1_k<<<gat_cdiv(NE * HEADS, B256), B256, 0, stream>>>(ei, asrc, adst, emax, edge_e, HEADS);
    gat_edge2_k<<<gat_cdiv(NE * HEADS, B256), B256, 0, stream>>>(ei, emax, edge_e, denom, HEADS);
    gat_edge3_k<<<gat_cdiv(NE * DLAY,  B256), B256, 0, stream>>>(ei, hW, DLAY, edge_e, denom, agg,
                                                                 HEADS, HID, DLAY);

    // + bias, BN, ELU -> act16[N,128]
    gat_ew_bn_k<<<gat_cdiv(NN * DLAY, B256), B256, 0, stream>>>(
        agg, DLAY, NN * DLAY, convB[l], bn_g[l], bn_b[l], bn_m[l], bn_v[l], 1, act16, nullptr);
  }

  // ---- output GAT layer (1 head x 8, no concat) ----
  // hW[N,16] = act16[N,128] @ outW[128,16]  (cols 8..15 zero-padded)
  gat_gemm_wmma_k<DLAY, 1><<<MT, 32, 0, stream>>>(act16, DLAY, woutP, hW, 16);

  gat_att_k<<<gat_cdiv(NN, B256), B256, 0, stream>>>(
      hW, 16, 1, NCLS, outAS, outAD, asrc, adst, NN);

  gat_fill_k<<<gat_cdiv(NN, B256),        B256, 0, stream>>>(emax,  -1e30f, NN);
  gat_fill_k<<<gat_cdiv(NN, B256),        B256, 0, stream>>>(denom,  0.0f,  NN);
  gat_fill_k<<<gat_cdiv(NN * NCLS, B256), B256, 0, stream>>>(agg,    0.0f,  NN * NCLS);

  gat_edge1_k<<<gat_cdiv(NE, B256),        B256, 0, stream>>>(ei, asrc, adst, emax, edge_e, 1);
  gat_edge2_k<<<gat_cdiv(NE, B256),        B256, 0, stream>>>(ei, emax, edge_e, denom, 1);
  gat_edge3_k<<<gat_cdiv(NE * NCLS, B256), B256, 0, stream>>>(ei, hW, 16, edge_e, denom, agg,
                                                              1, NCLS, NCLS);

  // ---- log_softmax -> d_out
  gat_logsoftmax_k<<<gat_cdiv(NN, B256), B256, 0, stream>>>(agg, outB, out);
}
